// MambaConvBlock_43276090474777
// MI455X (gfx1250) — matrix operand order repaired and checked by reading, not compiled, not run on hardware
//
#include <hip/hip_runtime.h>
#include <hip/hip_bf16.h>

typedef __attribute__((ext_vector_type(16))) _Float16 v16h;
typedef __attribute__((ext_vector_type(8)))  _Float16 v8h;
typedef __attribute__((ext_vector_type(8)))  float    v8f;

#define B_    2
#define HIN   64
#define CIN   128
#define HOUT  32
#define LSEQ  1024
#define MTOT  2048           // B_*LSEQ
#define DM    256
#define DI    512
#define DST   64
#define DTR   16
#define K3    1152           // CIN*9
#define NXP   144            // dt_rank + 2*d_state
#define NXPP  160            // NXP padded to multiple of 32

// ------------------------------------------------------------------
// elementwise / prep kernels
// ------------------------------------------------------------------
__global__ void k_cvt_f16(_Float16* __restrict__ dst, const float* __restrict__ src, int n) {
  int i = blockIdx.x * blockDim.x + threadIdx.x;
  if (i < n) dst[i] = (_Float16)src[i];
}

// convert with zero padding for elements >= n_src (used for padded x_proj weights)
__global__ void k_cvt_f16_pad(_Float16* __restrict__ dst, const float* __restrict__ src,
                              int n_src, int n_total) {
  int i = blockIdx.x * blockDim.x + threadIdx.x;
  if (i < n_total) dst[i] = (i < n_src) ? (_Float16)src[i] : (_Float16)0.f;
}

__global__ void k_bnprep(float* __restrict__ scale, float* __restrict__ bias,
                         const float* __restrict__ g, const float* __restrict__ b,
                         const float* __restrict__ m, const float* __restrict__ v, int n) {
  int i = blockIdx.x * blockDim.x + threadIdx.x;
  if (i < n) {
    float s = g[i] * rsqrtf(v[i] + 1e-5f);
    scale[i] = s;
    bias[i]  = b[i] - m[i] * s;
  }
}

__global__ void k_Aprep(float* __restrict__ Aexp, const float* __restrict__ Alog, int n) {
  int i = blockIdx.x * blockDim.x + threadIdx.x;
  if (i < n) Aexp[i] = -__expf(Alog[i]);
}

// im2col for conv3x3 stride2 pad1  ->  P3[MTOT][K3]  (f16)
__global__ void k_im2col3(_Float16* __restrict__ P3, const float* __restrict__ x) {
  int i = blockIdx.x * blockDim.x + threadIdx.x;
  if (i >= MTOT * K3) return;
  int m  = i / K3;
  int kk = i - m * K3;
  int c  = kk / 9;
  int r  = kk - c * 9;
  int kh = r / 3, kw = r - kh * 3;
  int b  = m >> 10;
  int l  = m & 1023;
  int oh = l >> 5, ow = l & 31;
  int ih = oh * 2 - 1 + kh;
  int iw = ow * 2 - 1 + kw;
  float v = 0.f;
  if (ih >= 0 && ih < HIN && iw >= 0 && iw < HIN)
    v = x[(((size_t)b * CIN + c) * HIN + ih) * HIN + iw];
  P3[i] = (_Float16)v;
}

// sampled input for 1x1 stride2 conv -> P1[MTOT][CIN] (f16)
__global__ void k_im2col1(_Float16* __restrict__ P1, const float* __restrict__ x) {
  int i = blockIdx.x * blockDim.x + threadIdx.x;
  if (i >= MTOT * CIN) return;
  int m = i >> 7;
  int c = i & 127;
  int b = m >> 10;
  int l = m & 1023;
  int oh = l >> 5, ow = l & 31;
  P1[i] = (_Float16)x[(((size_t)b * CIN + c) * HIN + oh * 2) * HIN + ow * 2];
}

// causal depthwise conv1d (k=4) + SiLU ; xs = first half of xz
__global__ void k_conv1d(float* __restrict__ xcf, _Float16* __restrict__ xch,
                         const float* __restrict__ xz,
                         const float* __restrict__ w, const float* __restrict__ bias) {
  int i = blockIdx.x * blockDim.x + threadIdx.x;       // (b*L + t)*DI + d
  if (i >= MTOT * DI) return;
  int d  = i & (DI - 1);
  int bt = i >> 9;
  int t  = bt & 1023;
  int b  = bt >> 10;
  float acc = bias[d];
#pragma unroll
  for (int k = 0; k < 4; k++) {
    int tt = t - 3 + k;
    if (tt >= 0)
      acc += w[d * 4 + k] * xz[((size_t)(b * LSEQ + tt)) * (2 * DI) + d];
  }
  float s = acc / (1.f + __expf(-acc));               // SiLU
  xcf[i] = s;
  xch[i] = (_Float16)s;
}

// dt = softplus(proj[:, :16] @ dt_proj_w^T + dt_proj_b)
__global__ void k_dt(float* __restrict__ dt, const float* __restrict__ proj,
                     const float* __restrict__ dtw, const float* __restrict__ dtb) {
  int i = blockIdx.x * blockDim.x + threadIdx.x;       // (b*L+t)*DI + d
  if (i >= MTOT * DI) return;
  int d  = i & (DI - 1);
  int bt = i >> 9;
  float acc = dtb[d];
  const float* pr = proj + (size_t)bt * NXPP;
#pragma unroll
  for (int r = 0; r < DTR; r++) acc += pr[r] * dtw[d * DTR + r];
  dt[i] = (acc > 20.f) ? acc : log1pf(__expf(acc));
}

// selective scan: one wave per (b, d); 2 states per lane (64 states / wave32)
__global__ void k_scan(_Float16* __restrict__ yact,
                       const float* __restrict__ dt, const float* __restrict__ xcf,
                       const float* __restrict__ proj, const float* __restrict__ xz,
                       const float* __restrict__ Aexp, const float* __restrict__ Dv) {
  int w    = (blockIdx.x * blockDim.x + threadIdx.x) >> 5;
  int lane = threadIdx.x & 31;
  if (w >= B_ * DI) return;
  int b = w >> 9;
  int d = w & (DI - 1);
  float A0 = Aexp[d * DST + lane];
  float A1 = Aexp[d * DST + lane + 32];
  float Dd = Dv[d];
  float h0 = 0.f, h1 = 0.f;
  for (int t = 0; t < LSEQ; t++) {
    size_t bt   = (size_t)b * LSEQ + t;
    float dtv = dt [bt * DI + d];
    float xcv = xcf[bt * DI + d];
    float zv  = xz [bt * (2 * DI) + DI + d];
    const float* pr = proj + bt * NXPP;
    float Bv0 = pr[16 + lane],  Bv1 = pr[16 + 32 + lane];
    float Cv0 = pr[80 + lane],  Cv1 = pr[80 + 32 + lane];
    float a0 = __expf(dtv * A0), a1 = __expf(dtv * A1);
    float btm = dtv * xcv;
    h0 = a0 * h0 + btm * Bv0;
    h1 = a1 * h1 + btm * Bv1;
    float p = h0 * Cv0 + h1 * Cv1;
#pragma unroll
    for (int off = 16; off > 0; off >>= 1) p += __shfl_xor(p, off, 32);
    if (lane == 0) {
      float y  = p + Dd * xcv;
      float sz = zv / (1.f + __expf(-zv));             // SiLU(z)
      yact[bt * DI + d] = (_Float16)(y * sz);
    }
  }
}

// ------------------------------------------------------------------
// generic f16 WMMA GEMM:  out[M,N] = A[M,K] * W[N,K]^T
// 32x32 tile per wave (2x2 WMMA tiles, full A/B fragment reuse).
// K pipelined 2-deep with two fixed buffer sets (no rotation copies).
// Requires M%32==0, N%32==0, K%64==0.
// modes: 0 plain f32 ; 1 BN+ReLU -> f16 ; 2 BN -> f32 ; 4 +ident,ReLU,NCHW -> dout
// ------------------------------------------------------------------
__device__ __forceinline__ void gemm_store(
    int m, int n, float val, int mode, int ldc,
    float* out0, _Float16* out1,
    const float* scale, const float* bias,
    const float* ident, float* dout) {
  if (mode == 0) {
    out0[(size_t)m * ldc + n] = val;
  } else if (mode == 1) {
    float t = val * scale[n] + bias[n];
    t = t > 0.f ? t : 0.f;
    out1[(size_t)m * ldc + n] = (_Float16)t;
  } else if (mode == 2) {
    float t = val * scale[n] + bias[n];
    out0[(size_t)m * ldc + n] = t;
  } else {                                             // final: relu(y + identity) -> NCHW
    float t = val + ident[(size_t)m * DM + n];
    t = t > 0.f ? t : 0.f;
    int b = m >> 10, l = m & 1023;
    dout[(size_t)b * (DM * LSEQ) + (size_t)n * LSEQ + l] = t;
  }
}

// A fragment (16x32 f16): lane m = lane&15 ; e<8 -> K=kgrp+e ; e>=8 -> K=kgrp+16+(e-8)
__device__ __forceinline__ v16h load_afrag(const _Float16* row, int k0, int kgrp) {
  v8h lo = *(const v8h*)(row + k0 + kgrp);
  v8h hi = *(const v8h*)(row + k0 + kgrp + 16);
  v16h f;
#pragma unroll
  for (int e = 0; e < 8; e++) { f[e] = lo[e]; f[e + 8] = hi[e]; }
  return f;
}

// B fragment (32x16 f16): lane n = lane&15 ; e<8 -> K=kbB+e ; e>=8 -> K=kbB+16+(e-8)
__device__ __forceinline__ v16h load_bfrag(const _Float16* row, int k0, int kbB) {
  v8h lo = *(const v8h*)(row + k0 + kbB);
  v8h hi = *(const v8h*)(row + k0 + kbB + 16);
  v16h f;
#pragma unroll
  for (int e = 0; e < 8; e++) { f[e] = lo[e]; f[e + 8] = hi[e]; }
  return f;
}

__device__ __forceinline__ void wmma4(const v16h& a0, const v16h& a1,
                                      const v16h& b0, const v16h& b1,
                                      v8f& c00, v8f& c01, v8f& c10, v8f& c11) {
  c00 = __builtin_amdgcn_wmma_f32_16x16x32_f16(false, a0, false, b0, (short)0, c00, false, false);
  c01 = __builtin_amdgcn_wmma_f32_16x16x32_f16(false, a0, false, b1, (short)0, c01, false, false);
  c10 = __builtin_amdgcn_wmma_f32_16x16x32_f16(false, a1, false, b0, (short)0, c10, false, false);
  c11 = __builtin_amdgcn_wmma_f32_16x16x32_f16(false, a1, false, b1, (short)0, c11, false, false);
}

__global__ void wmma_gemm_kernel(
    const _Float16* __restrict__ A, int lda,
    const _Float16* __restrict__ W, int ldw,
    int M, int N, int K,
    float* out0, _Float16* out1,
    const float* scale, const float* bias,
    const float* ident, float* dout,
    int ldc, int mode) {
  int wave = (blockIdx.x * blockDim.x + threadIdx.x) >> 5;
  int lane = threadIdx.x & 31;
  int nT = N >> 5;
  if (wave >= (M >> 5) * nT) return;
  int tm = wave / nT;
  int tn = wave - tm * nT;

  const int l15  = lane & 15;
  const int kgrp = (lane >> 4) * 8;                    // A: K group per half-wave
  const int kbB  = (lane >> 4) * 8;                    // B: K group per half-wave
  const _Float16* Arow0 = A + (size_t)(tm * 32 + l15) * lda;
  const _Float16* Arow1 = Arow0 + (size_t)16 * lda;
  const _Float16* Brow0 = W + (size_t)(tn * 32 + l15) * ldw;
  const _Float16* Brow1 = Brow0 + (size_t)16 * ldw;

  v8f acc00 = {}, acc01 = {}, acc10 = {}, acc11 = {};

  // two fixed fragment buffer sets; K pipelined 2-deep, no rotation copies
  v16h xa0 = load_afrag(Arow0, 0, kgrp);
  v16h xa1 = load_afrag(Arow1, 0, kgrp);
  v16h xb0 = load_bfrag(Brow0, 0, kbB);
  v16h xb1 = load_bfrag(Brow1, 0, kbB);
  v16h ya0, ya1, yb0, yb1;

  int k0 = 0;
  for (; k0 < K - 64; k0 += 64) {
    __builtin_prefetch(Arow0 + k0 + 96, 0, 1);
    __builtin_prefetch(Brow0 + k0 + 96, 0, 1);
    ya0 = load_afrag(Arow0, k0 + 32, kgrp);
    ya1 = load_afrag(Arow1, k0 + 32, kgrp);
    yb0 = load_bfrag(Brow0, k0 + 32, kbB);
    yb1 = load_bfrag(Brow1, k0 + 32, kbB);
    wmma4(xa0, xa1, xb0, xb1, acc00, acc01, acc10, acc11);
    xa0 = load_afrag(Arow0, k0 + 64, kgrp);
    xa1 = load_afrag(Arow1, k0 + 64, kgrp);
    xb0 = load_bfrag(Brow0, k0 + 64, kbB);
    xb1 = load_bfrag(Brow1, k0 + 64, kbB);
    wmma4(ya0, ya1, yb0, yb1, acc00, acc01, acc10, acc11);
  }
  // tail: last 64 of K
  ya0 = load_afrag(Arow0, k0 + 32, kgrp);
  ya1 = load_afrag(Arow1, k0 + 32, kgrp);
  yb0 = load_bfrag(Brow0, k0 + 32, kbB);
  yb1 = load_bfrag(Brow1, k0 + 32, kbB);
  wmma4(xa0, xa1, xb0, xb1, acc00, acc01, acc10, acc11);
  wmma4(ya0, ya1, yb0, yb1, acc00, acc01, acc10, acc11);

  // C/D layout: element r -> m = base + (lane>>4)*8 + r , n = base + (lane&15)
  const int mrA = tm * 32 + (lane >> 4) * 8;
  const int ncA = tn * 32 + l15;
#pragma unroll
  for (int r = 0; r < 8; r++) {
    gemm_store(mrA + r,      ncA,      acc00[r], mode, ldc, out0, out1, scale, bias, ident, dout);
    gemm_store(mrA + r,      ncA + 16, acc01[r], mode, ldc, out0, out1, scale, bias, ident, dout);
    gemm_store(mrA + r + 16, ncA,      acc10[r], mode, ldc, out0, out1, scale, bias, ident, dout);
    gemm_store(mrA + r + 16, ncA + 16, acc11[r], mode, ldc, out0, out1, scale, bias, ident, dout);
  }
}

// ------------------------------------------------------------------
extern "C" void kernel_launch(void* const* d_in, const int* in_sizes, int n_in,
                              void* d_out, int out_size, void* d_ws, size_t ws_size,
                              hipStream_t stream) {
  const float* x        = (const float*)d_in[0];
  const float* conv_w   = (const float*)d_in[1];
  const float* bn1_g    = (const float*)d_in[2];
  const float* bn1_b    = (const float*)d_in[3];
  const float* bn1_m    = (const float*)d_in[4];
  const float* bn1_v    = (const float*)d_in[5];
  const float* res_w    = (const float*)d_in[6];
  const float* bn2_g    = (const float*)d_in[7];
  const float* bn2_b    = (const float*)d_in[8];
  const float* bn2_m    = (const float*)d_in[9];
  const float* bn2_v    = (const float*)d_in[10];
  const float* in_proj  = (const float*)d_in[11];
  const float* c1d_w    = (const float*)d_in[12];
  const float* c1d_b    = (const float*)d_in[13];
  const float* x_proj   = (const float*)d_in[14];
  const float* dt_w     = (const float*)d_in[15];
  const float* dt_b     = (const float*)d_in[16];
  const float* A_log    = (const float*)d_in[17];
  const float* Dvec     = (const float*)d_in[18];
  const float* out_proj = (const float*)d_in[19];
  float* dout = (float*)d_out;

  // ---- workspace layout (256B aligned segments) ----
  size_t off = 0;
  auto alloc = [&](size_t bytes) -> char* {
    char* p = (char*)d_ws + off;
    off = (off + bytes + 255) & ~(size_t)255;
    return p;
  };
  _Float16* W1h  = (_Float16*)alloc((size_t)DM * K3 * 2);
  _Float16* Wrh  = (_Float16*)alloc((size_t)DM * CIN * 2);
  _Float16* Wip  = (_Float16*)alloc((size_t)(2 * DI) * DM * 2);
  _Float16* Wxp  = (_Float16*)alloc((size_t)NXPP * DI * 2);       // padded to 160 rows
  _Float16* Wop  = (_Float16*)alloc((size_t)DM * DI * 2);
  _Float16* P3   = (_Float16*)alloc((size_t)MTOT * K3 * 2);
  _Float16* P1   = (_Float16*)alloc((size_t)MTOT * CIN * 2);
  _Float16* u_h  = (_Float16*)alloc((size_t)MTOT * DM * 2);
  float*    iden = (float*)   alloc((size_t)MTOT * DM * 4);
  float*    xz   = (float*)   alloc((size_t)MTOT * 2 * DI * 4);
  float*    xcf  = (float*)   alloc((size_t)MTOT * DI * 4);
  _Float16* xch  = (_Float16*)alloc((size_t)MTOT * DI * 2);
  float*    proj = (float*)   alloc((size_t)MTOT * NXPP * 4);     // padded ldc = 160
  float*    dtv  = (float*)   alloc((size_t)MTOT * DI * 4);
  _Float16* yact = (_Float16*)alloc((size_t)MTOT * DI * 2);
  float*    Aexp = (float*)   alloc((size_t)DI * DST * 4);
  float*    s1   = (float*)   alloc(DM * 4);
  float*    b1   = (float*)   alloc(DM * 4);
  float*    s2   = (float*)   alloc(DM * 4);
  float*    b2   = (float*)   alloc(DM * 4);

  auto ew = [&](int n) { return dim3((n + 255) / 256); };

  // weight conversions + BN fold + A = -exp(A_log)
  k_cvt_f16<<<ew(DM * K3),     256, 0, stream>>>(W1h, conv_w,   DM * K3);
  k_cvt_f16<<<ew(DM * CIN),    256, 0, stream>>>(Wrh, res_w,    DM * CIN);
  k_cvt_f16<<<ew(2 * DI * DM), 256, 0, stream>>>(Wip, in_proj,  2 * DI * DM);
  k_cvt_f16_pad<<<ew(NXPP * DI), 256, 0, stream>>>(Wxp, x_proj, NXP * DI, NXPP * DI);
  k_cvt_f16<<<ew(DM * DI),     256, 0, stream>>>(Wop, out_proj, DM * DI);
  k_bnprep <<<1, 256, 0, stream>>>(s1, b1, bn1_g, bn1_b, bn1_m, bn1_v, DM);
  k_bnprep <<<1, 256, 0, stream>>>(s2, b2, bn2_g, bn2_b, bn2_m, bn2_v, DM);
  k_Aprep  <<<ew(DI * DST), 256, 0, stream>>>(Aexp, A_log, DI * DST);

  // im2col
  k_im2col3<<<ew(MTOT * K3),  256, 0, stream>>>(P3, x);
  k_im2col1<<<ew(MTOT * CIN), 256, 0, stream>>>(P1, x);

  auto gemm = [&](const _Float16* A, int lda, const _Float16* W, int ldw,
                  int M, int N, int K, float* o0, _Float16* o1,
                  const float* sc, const float* bi,
                  const float* id, float* dp, int ldc, int mode) {
    int waves = (M >> 5) * (N >> 5);
    wmma_gemm_kernel<<<(waves + 3) / 4, 128, 0, stream>>>(
        A, lda, W, ldw, M, N, K, o0, o1, sc, bi, id, dp, ldc, mode);
  };

  // conv3x3 + BN + ReLU -> u (f16, (B,L,256))
  gemm(P3, K3, W1h, K3, MTOT, DM, K3, nullptr, u_h, s1, b1, nullptr, nullptr, DM, 1);
  // residual 1x1 + BN -> identity (f32, (B,L,256))
  gemm(P1, CIN, Wrh, CIN, MTOT, DM, CIN, iden, nullptr, s2, b2, nullptr, nullptr, DM, 2);
  // in_proj -> xz (f32, (B,L,1024))
  gemm(u_h, DM, Wip, DM, MTOT, 2 * DI, DM, xz, nullptr, nullptr, nullptr, nullptr, nullptr, 2 * DI, 0);
  // depthwise causal conv1d + SiLU -> xc (f32+f16)
  k_conv1d<<<ew(MTOT * DI), 256, 0, stream>>>(xcf, xch, xz, c1d_w, c1d_b);
  // x_proj -> proj (f32, (B,L,160), cols 144..159 are padding)
  gemm(xch, DI, Wxp, DI, MTOT, NXPP, DI, proj, nullptr, nullptr, nullptr, nullptr, nullptr, NXPP, 0);
  // dt = softplus(dt_proj(proj[:, :16]))
  k_dt<<<ew(MTOT * DI), 256, 0, stream>>>(dtv, proj, dt_w, dt_b);
  // selective scan + gate -> yact (f16, (B,L,512))
  k_scan<<<(B_ * DI * 32) / 256, 256, 0, stream>>>(yact, dtv, xcf, proj, xz, Aexp, Dvec);
  // out_proj + identity + ReLU -> d_out (NCHW f32)
  gemm(yact, DI, Wop, DI, MTOT, DM, DI, nullptr, nullptr, nullptr, nullptr, iden, dout, DM, 4);
}
